// CrossAttention2_72198400245864
// MI455X (gfx1250) — matrix operand order repaired and checked by reading, not compiled, not run on hardware
//
#include <hip/hip_runtime.h>
#include <stdint.h>

// ---------------------------------------------------------------------------
// CDNA5 (gfx1250) wave32 WMMA cross-attention block.
//   f16 storage, f32 accumulation via V_WMMA_F32_16X16X32_F16.
//   - register double-buffered fragment loads (partial s_wait_loadcnt)
//   - uniform-base + 32-bit-offset + immediate addressing (saddr loads)
//   - rolled K loops (no full unroll -> no spills); attention runs 4-wave
//     blocks with launch_bounds(128,1) so its ~200-VGPR live set fits.
// ---------------------------------------------------------------------------

typedef __attribute__((ext_vector_type(16))) _Float16 v16h;
typedef __attribute__((ext_vector_type(8)))  float    v8f;

#define D_EMBED 1024
#define N_HEAD  16
#define DH      64
#define LQ      1024
#define SKV     3072
#define BATCH   2
#define LN_EPS  1e-5f
#define NEG_BIG (-1.0e30f)

union Frag16 { v16h v; uint4 u[2]; _Float16 h[16]; };
union H8     { _Float16 h[8]; uint4 u; };

// fragment load: uniform base pointer + 32-bit byte offsets
__device__ __forceinline__ v16h load_frag_o(const _Float16* base,
                                            uint32_t o0, uint32_t o1) {
    Frag16 f;
    f.u[0] = *(const uint4*)((const char*)base + o0);
    f.u[1] = *(const uint4*)((const char*)base + o1);
    return f.v;
}

__device__ __forceinline__ v16h load_frag_2x8(const _Float16* p0, const _Float16* p1) {
    Frag16 f;
    f.u[0] = *(const uint4*)p0;
    f.u[1] = *(const uint4*)p1;
    return f.v;
}

__device__ __forceinline__ v8f wmma_f16(v16h a, v16h b, v8f c) {
    // (neg_a, A, neg_b, B, c_mod, C, reuse_a, reuse_b)
    return __builtin_amdgcn_wmma_f32_16x16x32_f16(false, a, false, b, (short)0, c,
                                                  false, false);
}

// ---------------------------------------------------------------------------
// fp32 -> f16 conversion (grid-stride)
// ---------------------------------------------------------------------------
__global__ void f32_to_f16_kernel(const float* __restrict__ src,
                                  _Float16* __restrict__ dst, int n) {
    for (int i = blockIdx.x * blockDim.x + threadIdx.x; i < n;
         i += gridDim.x * blockDim.x)
        dst[i] = (_Float16)src[i];
}

// Fused concat(h_kv1, h_kv2) along seq -> f16 (B, S, D)
__global__ void build_hkv16_kernel(const float* __restrict__ kv1,
                                   const float* __restrict__ kv2,
                                   _Float16* __restrict__ dst) {
    const int n = BATCH * SKV * D_EMBED;
    for (int i = blockIdx.x * blockDim.x + threadIdx.x; i < n;
         i += gridDim.x * blockDim.x) {
        int c = i % D_EMBED;
        int r = (i / D_EMBED) % SKV;
        int b = i / (D_EMBED * SKV);
        float v = (r < LQ)
            ? kv1[((size_t)b * LQ + r) * D_EMBED + c]
            : kv2[((size_t)b * 2 * LQ + (r - LQ)) * D_EMBED + c];
        dst[i] = (_Float16)v;
    }
}

// ---------------------------------------------------------------------------
// GEMM: out = A (MxK, row-major f16) @ W^T (W is NxK row-major f16) + bias
// K = D_EMBED (compile-time). Rolled K loop (unroll 1) with explicit
// register double-buffering -> partial s_wait_loadcnt, no spills.
// Each wave: 16 rows x 64 cols. Block: 8 waves -> 32 rows x 256 cols.
// Grid: (N/256, M/32).
// TRANS=0: out[m*N + n]; TRANS=1: out[n*M + m] (dim-major, for V^T)
// ---------------------------------------------------------------------------
template <int TRANS>
__global__ __launch_bounds__(256, 1)
void gemm_f16_wmma(const _Float16* __restrict__ A,
                   const _Float16* __restrict__ W,
                   const float* __restrict__ bias,
                   _Float16* __restrict__ out,
                   int M, int N) {
    constexpr int K = D_EMBED;
    const int lane = threadIdx.x & 31;
    const int wave = threadIdx.x >> 5;
    const int hf = lane >> 4;      // half-wave select
    const int ln = lane & 15;
    const int rowBase = blockIdx.y * 32 + (wave >> 2) * 16;
    const int colBase = blockIdx.x * 256 + (wave & 3) * 64;

    v8f acc[4] = {};

    // 32-bit byte offsets from uniform bases (max ~13 MB, fits u32)
    const uint32_t aoff = (uint32_t)((rowBase + ln) * K + hf * 8) * 2u;
    const uint32_t boff = (uint32_t)((colBase + ln) * K + hf * 8) * 2u;

    auto loadA = [&](int kk) {
        return load_frag_o(A, aoff + kk * 2, aoff + (kk + 16) * 2);
    };
    auto loadB = [&](int t, int kk) {               // t*16*K*2: constant delta
        const uint32_t o = boff + (uint32_t)(t * 16 * K + kk) * 2u;
        return load_frag_o(W, o, o + 32);
    };

    // prologue: fill both buffers (K-steps 0 and 32)
    v16h a0 = loadA(0), a1 = loadA(32);
    v16h b0[4], b1[4];
#pragma unroll
    for (int t = 0; t < 4; ++t) { b0[t] = loadB(t, 0); b1[t] = loadB(t, 32); }

#pragma unroll 1
    for (int kk = 0; kk < K; kk += 64) {
        const int k2 = (kk + 64 < K) ? kk + 64 : 0;  // wrap wasted prefetch
        const int k3 = (kk + 96 < K) ? kk + 96 : 0;
#pragma unroll
        for (int t = 0; t < 4; ++t) acc[t] = wmma_f16(a0, b0[t], acc[t]);
        a0 = loadA(k2);
#pragma unroll
        for (int t = 0; t < 4; ++t) b0[t] = loadB(t, k2);
#pragma unroll
        for (int t = 0; t < 4; ++t) acc[t] = wmma_f16(a1, b1[t], acc[t]);
        a1 = loadA(k3);
#pragma unroll
        for (int t = 0; t < 4; ++t) b1[t] = loadB(t, k3);
    }

#pragma unroll
    for (int t = 0; t < 4; ++t) {
        const int col = colBase + t * 16 + ln;
        const float b = bias ? bias[col] : 0.0f;
        if (TRANS) {
            H8 tmp;
#pragma unroll
            for (int v = 0; v < 8; ++v)
                tmp.h[v] = (_Float16)(acc[t][v] + b);   // rows contiguous in v
            const uint32_t o = ((uint32_t)col * M + rowBase + hf * 8) * 2u;
            *(uint4*)((char*)out + o) = tmp.u;
        } else {
#pragma unroll
            for (int v = 0; v < 8; ++v) {
                const int row = rowBase + v + 8 * hf;
                *(_Float16*)((char*)out + ((uint32_t)row * N + col) * 2u) =
                    (_Float16)(acc[t][v] + b);
            }
        }
    }
}

// ---------------------------------------------------------------------------
// Flash-style attention. Q:(B*L,D) f16, K:(B*S,D) f16, Vt:(D, B*S) f16.
// One wave = one 16-query tile of one (b,h); online softmax over S in
// 32-key blocks. V loads issued before the QK^T WMMAs; next K block
// prefetched before the softmax VALU section. 4-wave blocks (one wave per
// SIMD) + launch_bounds(128,1): the full register file per wave, no spills.
// Grid: (L/64, H, B), 128 threads.
// ---------------------------------------------------------------------------
__global__ __launch_bounds__(128, 1)
void attention_wmma(const _Float16* __restrict__ Q,
                    const _Float16* __restrict__ Km,
                    const _Float16* __restrict__ Vt,
                    _Float16* __restrict__ ctx) {
    const int lane = threadIdx.x & 31;
    const int wave = threadIdx.x >> 5;
    const int hf = lane >> 4;
    const int ln = lane & 15;
    const int b = blockIdx.z;
    const int h = blockIdx.y;
    const int q0 = (blockIdx.x * 4 + wave) * 16;   // query base within batch
    const float scale = 0.125f;                    // 1/sqrt(64)
    const int BS = BATCH * SKV;
    const uint32_t ROWB = D_EMBED * 2;             // K-matrix row bytes

    __shared__ _Float16 pTile[4][16][32];          // per-wave P round-trip (4 KB)

    // Q fragments for K-dim 0..63 (two 16x32 A-frags)
    const uint32_t qoff =
        ((uint32_t)(b * LQ + q0 + ln) * D_EMBED + h * DH) * 2u + hf * 32;
    const v16h qf0 = load_frag_o(Q, qoff,      qoff + 16);
    const v16h qf1 = load_frag_o(Q, qoff + 64, qoff + 80);

    v8f acc[4] = {};           // ctx accumulator, 16 x 64
    float mrow[8], lrow[8];
#pragma unroll
    for (int v = 0; v < 8; ++v) { mrow[v] = NEG_BIG; lrow[v] = 0.0f; }

    const uint32_t koff =
        ((uint32_t)(b * SKV + ln) * D_EMBED + h * DH + hf * 16) * 2u;
    const uint32_t voff =
        ((uint32_t)(h * DH + ln) * BS + b * SKV + hf * 16) * 2u;

    auto loadK4 = [&](int s0, v16h* kf) {
        const uint32_t o = koff + (uint32_t)s0 * ROWB;
        kf[0] = load_frag_o(Km, o,                  o + 16);            // dims 0..31
        kf[1] = load_frag_o(Km, o + 64,             o + 80);            // dims 32..63
        kf[2] = load_frag_o(Km, o + 16 * ROWB,      o + 16 * ROWB + 16);
        kf[3] = load_frag_o(Km, o + 16 * ROWB + 64, o + 16 * ROWB + 80);
    };

    // Process one 32-key block using fragments kf; prefetch block sPre into kpre.
    auto process = [&](const v16h* kf, int s0, int sPre, v16h* kpre) {
        // V fragments issued early -> in flight across QK + softmax
        v16h vf[4];
        {
            const uint32_t o = voff + (uint32_t)s0 * 2u;
#pragma unroll
            for (int t = 0; t < 4; ++t) {
                const uint32_t ot = o + (uint32_t)(t * 16 * BS) * 2u;
                vf[t] = load_frag_o(Vt, ot, ot + 16);
            }
        }

        v8f sA = {};
        v8f sB = {};
        sA = wmma_f16(qf0, kf[0], sA); sA = wmma_f16(qf1, kf[1], sA);
        sB = wmma_f16(qf0, kf[2], sB); sB = wmma_f16(qf1, kf[3], sB);

        // prefetch next K block; overlaps the softmax VALU section below
        loadK4(sPre, kpre);

        // --- masked online softmax (row stats replicated in 16-lane groups) ---
        float e0[8], e1[8];
#pragma unroll
        for (int v = 0; v < 8; ++v) {
            const int lq = q0 + v + 8 * hf;        // query row within batch
            const int jA = s0 + ln;                // key indices of this lane
            const int jB = jA + 16;
            float a = sA[v] * scale;
            float c = sB[v] * scale;
            if ((jA & (LQ - 1)) > lq) a = NEG_BIG; // tripled causal: j % L > l
            if ((jB & (LQ - 1)) > lq) c = NEG_BIG;
            float mv = fmaxf(a, c);
            for (int off = 1; off < 16; off <<= 1)
                mv = fmaxf(mv, __shfl_xor(mv, off, 32));
            const float mnew = fmaxf(mrow[v], mv);
            const float alpha = __expf(mrow[v] - mnew);
            e0[v] = __expf(a - mnew);
            e1[v] = __expf(c - mnew);
            float rs = e0[v] + e1[v];
            for (int off = 1; off < 16; off <<= 1)
                rs += __shfl_xor(rs, off, 32);
            lrow[v] = lrow[v] * alpha + rs;
            mrow[v] = mnew;
#pragma unroll
            for (int t = 0; t < 4; ++t) acc[t][v] = acc[t][v] * alpha;
        }

        // --- P: C layout -> LDS -> A-fragment (16x32 f16) ---
#pragma unroll
        for (int v = 0; v < 8; ++v) {
            const int m = v + 8 * hf;
            pTile[wave][m][ln]      = (_Float16)e0[v];
            pTile[wave][m][16 + ln] = (_Float16)e1[v];
        }
        asm volatile("s_wait_dscnt 0" ::: "memory");
        const _Float16* Pr = &pTile[wave][ln][0];
        v16h pf = load_frag_2x8(Pr + hf * 16, Pr + hf * 16 + 8);

        // --- ctx += P * V ---
#pragma unroll
        for (int t = 0; t < 4; ++t) acc[t] = wmma_f16(pf, vf[t], acc[t]);
    };

    v16h kA[4], kB[4];
    loadK4(0, kA);
#pragma unroll 1
    for (int s0 = 0; s0 < SKV; s0 += 64) {
        process(kA, s0, s0 + 32, kB);
        const int sn = (s0 + 64 < SKV) ? s0 + 64 : 0;  // wrap wasted prefetch
        process(kB, s0 + 32, sn, kA);
    }

    // --- normalize and store ctx as f16 (B*L, D) ---
#pragma unroll
    for (int t = 0; t < 4; ++t) {
        const int col = h * DH + t * 16 + ln;
#pragma unroll
        for (int v = 0; v < 8; ++v) {
            const int row = b * LQ + q0 + v + 8 * hf;
            *(_Float16*)((char*)ctx + ((uint32_t)row * D_EMBED + col) * 2u) =
                (_Float16)(acc[t][v] / lrow[v]);
        }
    }
}

// ---------------------------------------------------------------------------
// Fused: out = LayerNorm(h_q + ctx @ Wo^T + out_b) * mask   (fp32 output)
// Block = 16-row strip x all 1024 cols (8 waves x 128 cols). Grid: M/16.
// Rolled K loop; A double-buffered, 8 B loads batched before the 8 WMMAs.
// ---------------------------------------------------------------------------
__global__ __launch_bounds__(256, 1)
void out_proj_ln_kernel(const _Float16* __restrict__ Ctx,
                        const _Float16* __restrict__ Wo,
                        const float* __restrict__ outb,
                        const float* __restrict__ hq,
                        const float* __restrict__ ln_g,
                        const float* __restrict__ ln_b,
                        const float* __restrict__ mask,
                        float* __restrict__ out) {
    constexpr int K = D_EMBED;
    const int lane = threadIdx.x & 31;
    const int wave = threadIdx.x >> 5;
    const int hf = lane >> 4;
    const int ln = lane & 15;
    const int rowBase = blockIdx.x * 16;
    const int colBase = wave * 128;

    __shared__ float s_sum[16], s_sq[16];
    if (threadIdx.x < 16) { s_sum[threadIdx.x] = 0.0f; s_sq[threadIdx.x] = 0.0f; }
    __syncthreads();

    v8f acc[8] = {};
    const uint32_t aoff = (uint32_t)((rowBase + ln) * K + hf * 8) * 2u;
    const uint32_t boff = (uint32_t)((colBase + ln) * K + hf * 8) * 2u;

    auto loadA = [&](int kk) {
        return load_frag_o(Ctx, aoff + kk * 2, aoff + (kk + 16) * 2);
    };
    auto loadB = [&](int t, int kk) {
        const uint32_t o = boff + (uint32_t)(t * 16 * K + kk) * 2u;
        return load_frag_o(Wo, o, o + 32);
    };

    v16h af0 = loadA(0), af1 = loadA(32);
#pragma unroll 1
    for (int kk = 0; kk < K; kk += 64) {
        const int k2 = (kk + 64 < K) ? kk + 64 : 0;
        const int k3 = (kk + 96 < K) ? kk + 96 : 0;
        v16h bf[8];
#pragma unroll
        for (int t = 0; t < 8; ++t) bf[t] = loadB(t, kk);
#pragma unroll
        for (int t = 0; t < 8; ++t) acc[t] = wmma_f16(af0, bf[t], acc[t]);
        af0 = loadA(k2);
#pragma unroll
        for (int t = 0; t < 8; ++t) bf[t] = loadB(t, kk + 32);
#pragma unroll
        for (int t = 0; t < 8; ++t) acc[t] = wmma_f16(af1, bf[t], acc[t]);
        af1 = loadA(k3);
    }

    // bias + residual; accumulate LN stats
    float psum[8], psq[8];
#pragma unroll
    for (int v = 0; v < 8; ++v) { psum[v] = 0.0f; psq[v] = 0.0f; }
#pragma unroll
    for (int t = 0; t < 8; ++t) {
        const int col = colBase + t * 16 + ln;
#pragma unroll
        for (int v = 0; v < 8; ++v) {
            const int row = rowBase + v + 8 * hf;
            const float* hp =
                (const float*)((const char*)hq + ((uint32_t)row * D_EMBED + col) * 4u);
            float y = acc[t][v] + outb[col] + *hp;
            acc[t][v] = y;
            psum[v] += y;
            psq[v]  += y * y;
        }
    }
#pragma unroll
    for (int v = 0; v < 8; ++v) {
        float s = psum[v], q = psq[v];
        for (int off = 1; off < 16; off <<= 1) {
            s += __shfl_xor(s, off, 32);
            q += __shfl_xor(q, off, 32);
        }
        if (ln == 0) {                          // one lane per 16-lane group
            atomicAdd(&s_sum[v + 8 * hf], s);   // ds_add_f32
            atomicAdd(&s_sq[v + 8 * hf], q);
        }
    }
    __syncthreads();

#pragma unroll
    for (int v = 0; v < 8; ++v) {
        const int r = v + 8 * hf;
        const float mu = s_sum[r] * (1.0f / D_EMBED);
        const float var = s_sq[r] * (1.0f / D_EMBED) - mu * mu;
        const float rstd = rsqrtf(var + LN_EPS);
        const int row = rowBase + r;
        const float mk = mask[row];
#pragma unroll
        for (int t = 0; t < 8; ++t) {
            const int col = colBase + t * 16 + ln;
            float o = (acc[t][v] - mu) * rstd * ln_g[col] + ln_b[col];
            *(float*)((char*)out + ((uint32_t)row * D_EMBED + col) * 4u) = o * mk;
        }
    }
}

// ---------------------------------------------------------------------------
extern "C" void kernel_launch(void* const* d_in, const int* in_sizes, int n_in,
                              void* d_out, int out_size, void* d_ws, size_t ws_size,
                              hipStream_t stream) {
    const float* h_q   = (const float*)d_in[0];
    const float* h_kv1 = (const float*)d_in[1];
    const float* h_kv2 = (const float*)d_in[2];
    const float* mask  = (const float*)d_in[3];
    const float* in_w  = (const float*)d_in[4];
    const float* in_b  = (const float*)d_in[5];
    const float* out_w = (const float*)d_in[6];
    const float* out_b = (const float*)d_in[7];
    const float* ln_g  = (const float*)d_in[8];
    const float* ln_b  = (const float*)d_in[9];

    const size_t nQ  = (size_t)BATCH * LQ * D_EMBED;    // 2M
    const size_t nKV = (size_t)BATCH * SKV * D_EMBED;   // 6M
    const size_t nW3 = (size_t)3 * D_EMBED * D_EMBED;   // 3M
    const size_t nW1 = (size_t)D_EMBED * D_EMBED;       // 1M

    _Float16* p      = (_Float16*)d_ws;
    _Float16* hq16   = p; p += nQ;
    _Float16* hkv16  = p; p += nKV;
    _Float16* wqkv16 = p; p += nW3;
    _Float16* wo16   = p; p += nW1;
    _Float16* q16    = p; p += nQ;
    _Float16* k16    = p; p += nKV;
    _Float16* vt16   = p; p += nKV;   // (D, B*S) dim-major
    _Float16* ctx16  = p; p += nQ;

    // conversions
    f32_to_f16_kernel<<<2048, 256, 0, stream>>>(h_q, hq16, (int)nQ);
    build_hkv16_kernel<<<2048, 256, 0, stream>>>(h_kv1, h_kv2, hkv16);
    f32_to_f16_kernel<<<2048, 256, 0, stream>>>(in_w, wqkv16, (int)nW3);
    f32_to_f16_kernel<<<2048, 256, 0, stream>>>(out_w, wo16, (int)nW1);

    // projections: grid (N/256, M/32)
    const int MQ = BATCH * LQ;    // 2048
    const int MK = BATCH * SKV;   // 6144
    gemm_f16_wmma<0><<<dim3(D_EMBED / 256, MQ / 32), 256, 0, stream>>>(
        hq16, wqkv16, in_b, q16, MQ, D_EMBED);
    gemm_f16_wmma<0><<<dim3(D_EMBED / 256, MK / 32), 256, 0, stream>>>(
        hkv16, wqkv16 + (size_t)D_EMBED * D_EMBED, in_b + D_EMBED,
        k16, MK, D_EMBED);
    gemm_f16_wmma<1><<<dim3(D_EMBED / 256, MK / 32), 256, 0, stream>>>(
        hkv16, wqkv16 + (size_t)2 * D_EMBED * D_EMBED, in_b + 2 * D_EMBED,
        vt16, MK, D_EMBED);

    // attention: (L/64, H, B), 4 waves per block
    attention_wmma<<<dim3(LQ / 64, N_HEAD, BATCH), 128, 0, stream>>>(
        q16, k16, vt16, ctx16);

    // out-proj + residual + LN + mask
    out_proj_ln_kernel<<<MQ / 16, 256, 0, stream>>>(
        ctx16, wo16, out_b, h_q, ln_g, ln_b, mask, (float*)d_out);
}
